// MSARowAttention_481036337737
// MI455X (gfx1250) — matrix operand order repaired and checked by reading, not compiled, not run on hardware
//
#include <hip/hip_runtime.h>
#include <stdint.h>

// ---------------------------------------------------------------------------
// MSA row attention for MI455X (gfx1250), wave32 + WMMA bf16 16x16x32 + TDM.
// B=1, N=128, L=256, D=256, DP=128, H=8, DH=32.
// ---------------------------------------------------------------------------

typedef __attribute__((ext_vector_type(16))) __bf16 v16bf;
typedef __attribute__((ext_vector_type(8)))  float  v8f;

// TDM descriptor group vector types (per probe: g0=uint32x4, g1=int32x8, g2/g3=int32x4)
typedef __attribute__((ext_vector_type(4))) unsigned int tdm_v4u;
typedef __attribute__((ext_vector_type(8))) int          tdm_v8i;
typedef __attribute__((ext_vector_type(4))) int          tdm_v4i;

// ROCm 7.2 clang-22 -> 5-arg tensor_load builtin; amdgpu-toolchain (therock-10.0
// headers present) -> 6-arg form. Discriminate via the TDM header shipping only
// with the latter.
#if __has_include(<hip/amd_detail/amd_gfx1250_TDM.h>)
#define TDM_SIX_ARG 1
#else
#define TDM_SIX_ARG 0
#endif

struct U4 { unsigned int x, y, z, w; };
union V16 { v16bf v; U4 u[2]; };

__device__ __forceinline__ unsigned short f2bf(float f) {
  unsigned int x = __float_as_uint(f);
  x += 0x7FFFu + ((x >> 16) & 1u);          // round-to-nearest-even
  return (unsigned short)(x >> 16);
}
__device__ __forceinline__ unsigned int pk2(float a, float b) {
  return (unsigned int)f2bf(a) | ((unsigned int)f2bf(b) << 16);
}
// load 8 consecutive fp32 and pack to 8 bf16 (one 16-byte A-frag chunk)
__device__ __forceinline__ U4 ld8f_bf(const float* p) {
  U4 r;
  r.x = pk2(p[0], p[1]); r.y = pk2(p[2], p[3]);
  r.z = pk2(p[4], p[5]); r.w = pk2(p[6], p[7]);
  return r;
}
__device__ __forceinline__ v8f wmma_bf16(v16bf a, v16bf b, v8f c) {
  return __builtin_amdgcn_wmma_f32_16x16x32_bf16(false, a, false, b,
                                                 (short)0, c, false, false);
}

// ---------------------------------------------------------------------------
// TDM: async 2D tile load Global -> LDS (ISA 08_async_tensor §8).
// dims/strides in elements of data_size=2 bytes. Issued by one wave; fence
// with s_wait_tensorcnt before consuming the LDS tile.
// ---------------------------------------------------------------------------
__device__ __forceinline__ void tdm_load_tile_2d(
    unsigned lds_off_bytes, unsigned long long gaddr_bytes,
    unsigned tensor_d0, unsigned tensor_d1,
    unsigned tile_d0, unsigned tile_d1, unsigned stride0) {
  tdm_v4u g0;
  g0[0] = 1u;                                           // count=1 (valid D#)
  g0[1] = lds_off_bytes;                                // lds_addr
  g0[2] = (unsigned)(gaddr_bytes & 0xFFFFFFFFu);        // global_addr[31:0]
  g0[3] = (unsigned)((gaddr_bytes >> 32) & 0x01FFFFFFu) // global_addr[56:32]
          | 0x80000000u;                                // type=2 ("image")
  tdm_v8i g1;
  g1[0] = (int)(1u << 16);                              // data_size=1 -> 2 bytes
  g1[1] = (int)((tensor_d0 & 0xFFFFu) << 16);           // tensor_dim0[15:0]
  g1[2] = (int)(((tensor_d0 >> 16) & 0xFFFFu) |         // tensor_dim0[31:16]
                ((tensor_d1 & 0xFFFFu) << 16));         // tensor_dim1[15:0]
  g1[3] = (int)(((tensor_d1 >> 16) & 0xFFFFu) |         // tensor_dim1[31:16]
                ((tile_d0 & 0xFFFFu) << 16));           // tile_dim0
  g1[4] = (int)(tile_d1 & 0xFFFFu);                     // tile_dim1 (tile_dim2=0)
  g1[5] = (int)stride0;                                 // tensor_dim0_stride[31:0]
  g1[6] = 0;                                            // stride0[47:32] | stride1 lo
  g1[7] = 0;
  tdm_v4i z4 = {0, 0, 0, 0};
#if TDM_SIX_ARG
  tdm_v8i z8 = {0, 0, 0, 0, 0, 0, 0, 0};
  __builtin_amdgcn_tensor_load_to_lds(g0, g1, z4, z4, z8, 0);
#else
  __builtin_amdgcn_tensor_load_to_lds(g0, g1, z4, z4, 0);
#endif
}

// ---------------------------------------------------------------------------
// Weight prep: W[K][N] fp32 -> Wt[Npad][K] bf16 (transposed, zero-padded cols)
// ---------------------------------------------------------------------------
__global__ void prep_w(const float* __restrict__ W, __bf16* __restrict__ Wt,
                       int K, int N, int Npad) {
  int idx = blockIdx.x * blockDim.x + threadIdx.x;
  if (idx >= Npad * K) return;
  int n = idx / K, k = idx - n * K;
  float v = (n < N) ? W[(size_t)k * N + n] : 0.f;
  *(unsigned short*)&Wt[idx] = f2bf(v);
}

// ---------------------------------------------------------------------------
// LayerNorm over D=256, one row per 256-thread block, bf16 output
// ---------------------------------------------------------------------------
__global__ __launch_bounds__(256) void ln_kernel(
    const float* __restrict__ msa, const float* __restrict__ g,
    const float* __restrict__ b, __bf16* __restrict__ xn) {
  const int row = blockIdx.x, tid = threadIdx.x;
  const int wave = tid >> 5, lane = tid & 31;
  float x = msa[(size_t)row * 256 + tid];
  float s = x, s2 = x * x;
  #pragma unroll
  for (int m = 16; m; m >>= 1) {
    s  += __shfl_xor(s, m, 32);
    s2 += __shfl_xor(s2, m, 32);
  }
  __shared__ float sh[16];
  if (lane == 0) { sh[wave] = s; sh[8 + wave] = s2; }
  __syncthreads();
  if (wave == 0) {
    float a = (lane < 8) ? sh[lane] : 0.f;
    float c = (lane < 8) ? sh[8 + lane] : 0.f;
    #pragma unroll
    for (int m = 4; m; m >>= 1) {
      a += __shfl_xor(a, m, 32);
      c += __shfl_xor(c, m, 32);
    }
    if (lane == 0) { sh[0] = a; sh[1] = c; }
  }
  __syncthreads();
  const float mean = sh[0] * (1.f / 256.f);
  const float var  = sh[1] * (1.f / 256.f) - mean * mean;
  const float rstd = rsqrtf(var + 1e-5f);
  const float y = (x - mean) * rstd * g[tid] + b[tid];
  *(unsigned short*)&xn[(size_t)row * 256 + tid] = f2bf(y);
}

// ---------------------------------------------------------------------------
// Projection GEMM: C[M=32768, 256] = A[M,256](bf16) @ Wt^T + bias
// mode 0: bf16 store (q/k/v); mode 1: fp32 sigmoid store (gate)
// one 16x16 tile per wave, 8 waves/block
// ---------------------------------------------------------------------------
__global__ __launch_bounds__(256) void gemm_proj(
    const __bf16* __restrict__ A, const __bf16* __restrict__ Wt,
    const float* __restrict__ bias, __bf16* __restrict__ Cb,
    float* __restrict__ Cf, const int mode) {
  const int lane = threadIdx.x & 31, wave = threadIdx.x >> 5;
  const int half = lane >> 4, l16 = lane & 15;
  const int t = blockIdx.x * 8 + wave;
  const int tn = t & 15, tm = t >> 4;
  const int m0 = tm * 16, n0 = tn * 16;

  const __bf16* ap = A  + (size_t)(m0 + l16) * 256;
  const __bf16* bp = Wt + (size_t)(n0 + l16) * 256 + half * 16;
  __builtin_prefetch(ap, 0, 0);     // global_prefetch_b8
  __builtin_prefetch(bp, 0, 0);
  v8f acc = {0.f, 0.f, 0.f, 0.f, 0.f, 0.f, 0.f, 0.f};
  #pragma unroll
  for (int k0 = 0; k0 < 256; k0 += 32) {
    V16 a, b;
    a.u[0] = *(const U4*)(ap + k0 + half * 16);
    a.u[1] = *(const U4*)(ap + k0 + half * 16 + 8);
    b.u[0] = *(const U4*)(bp + k0);
    b.u[1] = *(const U4*)(bp + k0 + 8);
    acc = wmma_bf16(a.v, b.v, acc);
  }
  const int nn = n0 + l16;
  const float bs = bias[nn];
  #pragma unroll
  for (int r = 0; r < 8; ++r) {
    const size_t m = (size_t)(m0 + half * 8 + r);
    const float c = acc[r] + bs;
    if (mode == 0) *(unsigned short*)&Cb[m * 256 + nn] = f2bf(c);
    else           Cf[m * 256 + nn] = 1.0f / (1.0f + __expf(-c));
  }
}

// ---------------------------------------------------------------------------
// Pair bias GEMM: bias[H][L][L] = pair[65536,128] @ Wb(pad 16) + bb
// A converted fp32->bf16 on the fly
// ---------------------------------------------------------------------------
__global__ __launch_bounds__(256) void biasproj_kernel(
    const float* __restrict__ pair, const __bf16* __restrict__ Wbt,
    const float* __restrict__ bb, float* __restrict__ biasHL) {
  const int lane = threadIdx.x & 31, wave = threadIdx.x >> 5;
  const int half = lane >> 4, l16 = lane & 15;
  const int m0 = (blockIdx.x * 8 + wave) * 16;

  const float* ar = pair + (size_t)(m0 + l16) * 128;
  v8f acc = {0.f, 0.f, 0.f, 0.f, 0.f, 0.f, 0.f, 0.f};
  #pragma unroll
  for (int k0 = 0; k0 < 128; k0 += 32) {
    V16 a, b;
    a.u[0] = ld8f_bf(ar + k0 + half * 16);
    a.u[1] = ld8f_bf(ar + k0 + half * 16 + 8);
    const __bf16* bp = Wbt + (size_t)l16 * 128 + k0 + half * 16;
    b.u[0] = *(const U4*)bp;
    b.u[1] = *(const U4*)(bp + 8);
    acc = wmma_bf16(a.v, b.v, acc);
  }
  const int h = l16;
  if (h < 8) {
    const float bbv = bb[h];
    #pragma unroll
    for (int r = 0; r < 8; ++r) {
      const size_t m = (size_t)(m0 + half * 8 + r);
      biasHL[(size_t)h * 65536 + m] = acc[r] + bbv;
    }
  }
}

// ---------------------------------------------------------------------------
// Fused attention per (n,h): S = scale*Q@K^T + bias + mask; softmax; O = P@V
// 8 waves/block, each wave handles two 16-row query tiles.
// K tile staged in LDS by the Tensor Data Mover (async DMA, TENSORcnt fence);
// V staged transposed by the VALU path in parallel.
// ---------------------------------------------------------------------------
__global__ __launch_bounds__(256) void attn_kernel(
    const __bf16* __restrict__ Q, const __bf16* __restrict__ Kmat,
    const __bf16* __restrict__ Vmat, const float* __restrict__ biasHL,
    const int* __restrict__ mask, __bf16* __restrict__ O) {
  const int n = blockIdx.x >> 3, h = blockIdx.x & 7;
  const int lane = threadIdx.x & 31, wave = threadIdx.x >> 5;
  const int half = lane >> 4, l16 = lane & 15;

  __shared__ __bf16 Kl[256 * 32];       // [j][dh]  <- filled by TDM
  __shared__ __bf16 Vt[32 * 256];       // [dh][j] (transposed for B-frags)
  __shared__ __bf16 Pbuf[8][16 * 32];   // per-wave P re-layout bounce

  const size_t base = (size_t)n * 256 * 256 + (size_t)h * 32;

  // --- async TDM load of the K tile: 256 rows x 32 elems, row stride 256 ---
  if (wave == 0) {
    const unsigned kl_off = (unsigned)(size_t)(&Kl[0]);   // LDS byte offset
    const unsigned long long ga = (unsigned long long)(size_t)(Kmat + base);
    tdm_load_tile_2d(kl_off, ga, /*tensor_d0=*/256, /*tensor_d1=*/256,
                     /*tile_d0=*/32, /*tile_d1=*/256, /*stride0=*/256);
  }
  // --- meanwhile: stage V transposed with the vector path ---
  for (int idx = threadIdx.x; idx < 256 * 32; idx += 256) {
    const int j = idx >> 5, dh = idx & 31;
    Vt[dh * 256 + j] = Vmat[base + (size_t)j * 256 + dh];
  }
  if (wave == 0) __builtin_amdgcn_s_wait_tensorcnt(0);    // K tile landed
  __syncthreads();

  const float scale = 0.17677669529663687f;  // 1/sqrt(32)

  for (int qi = 0; qi < 2; ++qi) {
    const int q0 = (qi * 8 + wave) * 16;
    const __bf16* qp = Q + base + (size_t)(q0 + l16) * 256;
    V16 qa;
    qa.u[0] = *(const U4*)(qp + half * 16);
    qa.u[1] = *(const U4*)(qp + half * 16 + 8);

    // S row-block: 16 rows x 256 keys, 16 accumulator tiles
    v8f s[16];
    #pragma unroll
    for (int jt = 0; jt < 16; ++jt) {
      V16 kb;
      const __bf16* kp = Kl + (jt * 16 + l16) * 32 + half * 16;
      kb.u[0] = *(const U4*)kp;
      kb.u[1] = *(const U4*)(kp + 8);
      v8f z = {0.f, 0.f, 0.f, 0.f, 0.f, 0.f, 0.f, 0.f};
      s[jt] = wmma_bf16(qa.v, kb.v, z);   // DH=32 == K: one WMMA per tile
    }

    // scale + pair bias + key mask
    const float* bp = biasHL + ((size_t)h * 256 + q0) * 256;
    #pragma unroll
    for (int jt = 0; jt < 16; ++jt) {
      const int j = jt * 16 + l16;
      const float mz = mask[n * 256 + j] ? 0.f : -1e9f;
      #pragma unroll
      for (int r = 0; r < 8; ++r) {
        const int i = half * 8 + r;
        s[jt][r] = s[jt][r] * scale + bp[(size_t)i * 256 + j] + mz;
      }
    }

    // exact softmax: row reduce across 16 lanes of each half-wave
    float rinv[8], rmax[8];
    #pragma unroll
    for (int r = 0; r < 8; ++r) {
      float m = s[0][r];
      #pragma unroll
      for (int jt = 1; jt < 16; ++jt) m = fmaxf(m, s[jt][r]);
      m = fmaxf(m, __shfl_xor(m, 1, 32));
      m = fmaxf(m, __shfl_xor(m, 2, 32));
      m = fmaxf(m, __shfl_xor(m, 4, 32));
      m = fmaxf(m, __shfl_xor(m, 8, 32));
      rmax[r] = m;
    }
    #pragma unroll
    for (int r = 0; r < 8; ++r) {
      float acc = 0.f;
      #pragma unroll
      for (int jt = 0; jt < 16; ++jt) {
        const float p = __expf(s[jt][r] - rmax[r]);
        s[jt][r] = p;
        acc += p;
      }
      acc += __shfl_xor(acc, 1, 32);
      acc += __shfl_xor(acc, 2, 32);
      acc += __shfl_xor(acc, 4, 32);
      acc += __shfl_xor(acc, 8, 32);
      rinv[r] = 1.0f / acc;
    }

    // O = P @ V over 8 chunks of 32 keys (LDS bounce: C-layout -> A-frag)
    v8f o0 = {0.f, 0.f, 0.f, 0.f, 0.f, 0.f, 0.f, 0.f};
    v8f o1 = {0.f, 0.f, 0.f, 0.f, 0.f, 0.f, 0.f, 0.f};
    for (int kc = 0; kc < 8; ++kc) {
      #pragma unroll
      for (int t = 0; t < 2; ++t) {
        const int jt = 2 * kc + t;
        #pragma unroll
        for (int r = 0; r < 8; ++r) {
          const int row = half * 8 + r, col = t * 16 + l16;
          *(unsigned short*)&Pbuf[wave][row * 32 + col] = f2bf(s[jt][r] * rinv[r]);
        }
      }
      __builtin_amdgcn_wave_barrier();   // per-wave LDS, DS ops in-order
      V16 pa;
      const __bf16* pp = &Pbuf[wave][l16 * 32];
      pa.u[0] = *(const U4*)(pp + half * 16);
      pa.u[1] = *(const U4*)(pp + half * 16 + 8);
      const __bf16* vb0 = Vt + (size_t)l16 * 256 + kc * 32 + half * 16;
      const __bf16* vb1 = Vt + (size_t)(16 + l16) * 256 + kc * 32 + half * 16;
      V16 b0, b1;
      b0.u[0] = *(const U4*)vb0; b0.u[1] = *(const U4*)(vb0 + 8);
      b1.u[0] = *(const U4*)vb1; b1.u[1] = *(const U4*)(vb1 + 8);
      o0 = wmma_bf16(pa.v, b0.v, o0);
      o1 = wmma_bf16(pa.v, b1.v, o1);
      __builtin_amdgcn_wave_barrier();
    }

    __bf16* op = O + base + (size_t)q0 * 256;
    #pragma unroll
    for (int r = 0; r < 8; ++r) {
      const int i = half * 8 + r;
      *(unsigned short*)&op[(size_t)i * 256 + l16]      = f2bf(o0[r]);
      *(unsigned short*)&op[(size_t)i * 256 + 16 + l16] = f2bf(o1[r]);
    }
  }
}

// ---------------------------------------------------------------------------
// Output GEMM: out = (attn_out @ Wo + bo) * gate, fp32 store
// ---------------------------------------------------------------------------
__global__ __launch_bounds__(256) void gemm_out(
    const __bf16* __restrict__ A, const __bf16* __restrict__ Wt,
    const float* __restrict__ bias, const float* __restrict__ gate,
    float* __restrict__ out) {
  const int lane = threadIdx.x & 31, wave = threadIdx.x >> 5;
  const int half = lane >> 4, l16 = lane & 15;
  const int t = blockIdx.x * 8 + wave;
  const int tn = t & 15, tm = t >> 4;
  const int m0 = tm * 16, n0 = tn * 16;

  const __bf16* ap = A  + (size_t)(m0 + l16) * 256;
  const __bf16* bp = Wt + (size_t)(n0 + l16) * 256 + half * 16;
  v8f acc = {0.f, 0.f, 0.f, 0.f, 0.f, 0.f, 0.f, 0.f};
  #pragma unroll
  for (int k0 = 0; k0 < 256; k0 += 32) {
    V16 a, b;
    a.u[0] = *(const U4*)(ap + k0 + half * 16);
    a.u[1] = *(const U4*)(ap + k0 + half * 16 + 8);
    b.u[0] = *(const U4*)(bp + k0);
    b.u[1] = *(const U4*)(bp + k0 + 8);
    acc = wmma_bf16(a.v, b.v, acc);
  }
  const int nn = n0 + l16;
  const float bs = bias[nn];
  #pragma unroll
  for (int r = 0; r < 8; ++r) {
    const size_t m = (size_t)(m0 + half * 8 + r);
    out[m * 256 + nn] = (acc[r] + bs) * gate[m * 256 + nn];
  }
}

// ---------------------------------------------------------------------------
extern "C" void kernel_launch(void* const* d_in, const int* in_sizes, int n_in,
                              void* d_out, int out_size, void* d_ws, size_t ws_size,
                              hipStream_t stream) {
  (void)in_sizes; (void)n_in; (void)out_size; (void)ws_size;
  const float* msa  = (const float*)d_in[0];
  const float* pair = (const float*)d_in[1];
  const int*   mask = (const int*)d_in[2];
  const float* ln_g = (const float*)d_in[3];
  const float* ln_b = (const float*)d_in[4];
  const float* Wq = (const float*)d_in[5];  const float* bq = (const float*)d_in[6];
  const float* Wk = (const float*)d_in[7];  const float* bk = (const float*)d_in[8];
  const float* Wv = (const float*)d_in[9];  const float* bv = (const float*)d_in[10];
  const float* Wb = (const float*)d_in[11]; const float* bb = (const float*)d_in[12];
  const float* Wo = (const float*)d_in[13]; const float* bo = (const float*)d_in[14];
  const float* Wg = (const float*)d_in[15]; const float* bg = (const float*)d_in[16];

  char* ws = (char*)d_ws;
  const size_t SZB = (size_t)32768 * 256 * 2;   // one bf16 [32768,256] buffer
  size_t off = 0;
  __bf16* xn   = (__bf16*)(ws + off); off += SZB;
  __bf16* qb   = (__bf16*)(ws + off); off += SZB;
  __bf16* kb   = (__bf16*)(ws + off); off += SZB;
  __bf16* vb   = (__bf16*)(ws + off); off += SZB;
  __bf16* ao   = (__bf16*)(ws + off); off += SZB;
  float*  gate = (float*)(ws + off);  off += (size_t)32768 * 256 * 4;
  float*  biasHL = (float*)(ws + off); off += (size_t)8 * 256 * 256 * 4;
  __bf16* Wqt = (__bf16*)(ws + off); off += (size_t)256 * 256 * 2;
  __bf16* Wkt = (__bf16*)(ws + off); off += (size_t)256 * 256 * 2;
  __bf16* Wvt = (__bf16*)(ws + off); off += (size_t)256 * 256 * 2;
  __bf16* Wot = (__bf16*)(ws + off); off += (size_t)256 * 256 * 2;
  __bf16* Wgt = (__bf16*)(ws + off); off += (size_t)256 * 256 * 2;
  __bf16* Wbt = (__bf16*)(ws + off); off += (size_t)16 * 128 * 2;

  // 1) weight prep (transpose + bf16)
  prep_w<<<256, 256, 0, stream>>>(Wq, Wqt, 256, 256, 256);
  prep_w<<<256, 256, 0, stream>>>(Wk, Wkt, 256, 256, 256);
  prep_w<<<256, 256, 0, stream>>>(Wv, Wvt, 256, 256, 256);
  prep_w<<<256, 256, 0, stream>>>(Wo, Wot, 256, 256, 256);
  prep_w<<<256, 256, 0, stream>>>(Wg, Wgt, 256, 256, 256);
  prep_w<<<8,   256, 0, stream>>>(Wb, Wbt, 128, 8, 16);

  // 2) LayerNorm -> xn (bf16)
  ln_kernel<<<32768, 256, 0, stream>>>(msa, ln_g, ln_b, xn);

  // 3) Q/K/V + gate projections (WMMA)
  gemm_proj<<<4096, 256, 0, stream>>>(xn, Wqt, bq, qb, (float*)nullptr, 0);
  gemm_proj<<<4096, 256, 0, stream>>>(xn, Wkt, bk, kb, (float*)nullptr, 0);
  gemm_proj<<<4096, 256, 0, stream>>>(xn, Wvt, bv, vb, (float*)nullptr, 0);
  gemm_proj<<<4096, 256, 0, stream>>>(xn, Wgt, bg, (__bf16*)nullptr, gate, 1);

  // 4) pair bias projection (WMMA, N padded to 16)
  biasproj_kernel<<<512, 256, 0, stream>>>(pair, Wbt, bb, biasHL);

  // 5) fused attention, one block per (n,h), K tile via TDM
  attn_kernel<<<128 * 8, 256, 0, stream>>>(qb, kb, vb, biasHL, mask, ao);

  // 6) output projection * gate -> fp32 result
  gemm_out<<<4096, 256, 0, stream>>>(ao, Wot, bo, gate, (float*)d_out);
}
